// MLPPredictor_84335977824414
// MI455X (gfx1250) — compile-verified, matrix operand order checked by reading, not run on hardware
//
#include <hip/hip_runtime.h>

typedef __attribute__((ext_vector_type(16))) _Float16 v16h;
typedef __attribute__((ext_vector_type(4)))  _Float16 v4h;
typedef __attribute__((ext_vector_type(8)))  float    v8f;

union AF { v16h v; unsigned u[8]; };

#define WAVES 8
#define TPB   (WAVES * 32)

// Load the four 16x32 f16 A-fragments (K = 0..127) for this lane's row from a
// row-major [16][128] f16 LDS tile. Pairs along K are contiguous -> u32 loads.
// Layout per ISA 7.12.2: lanes 0-15 hold K blocks {0-7,16-23}+kb*32,
// lanes 16-31 hold {8-15,24-31}+kb*32.
__device__ __forceinline__ void load_afrags(AF* f, const _Float16* row_base, int hi) {
    const unsigned* s = (const unsigned*)row_base;  // 64 u32 per 128-half row
#pragma unroll
    for (int kb = 0; kb < 4; ++kb)
#pragma unroll
        for (int j = 0; j < 8; ++j)
            f[kb].u[j] = s[kb * 16 + hi * 4 + ((j >> 2) << 3) + (j & 3)];
}

// Load one 32x16 B-fragment for K-block kb from a transposed weight row
// Wt[n][0..127] (so pairs along K are contiguous). Same K striping as A.
__device__ __forceinline__ v16h load_bfrag(const _Float16* wrow, int kb, int hi) {
    AF b;
    const unsigned* s = (const unsigned*)wrow;
#pragma unroll
    for (int j = 0; j < 8; ++j)
        b.u[j] = s[kb * 16 + hi * 4 + ((j >> 2) << 3) + (j & 3)];
    return b.v;
}

// One 128->128 layer with ReLU: H(16x128,f16 LDS) = relu(H @ Wt^T-stored).
// A-fragments are hoisted to registers, 8 N-tile accumulators live at once,
// then results overwrite the same H tile (all H reads already done).
__device__ __forceinline__ void layer128(_Float16* Hw, const _Float16* Wt,
                                         int row, int hi) {
    AF a[4];
    load_afrags(a, Hw + row * 128, hi);

    v8f acc[8] = {};
#pragma unroll
    for (int nb = 0; nb < 8; ++nb) {
        const _Float16* wrow = Wt + (nb * 16 + row) * 128;
#pragma unroll
        for (int kb = 0; kb < 4; ++kb) {
            v16h b = load_bfrag(wrow, kb, hi);
            acc[nb] = __builtin_amdgcn_wmma_f32_16x16x32_f16(
                false, a[kb].v, false, b, (short)0, acc[nb], false, false);
        }
    }

    // D layout: n = lane&15, m = r + 8*(lane>=16). ReLU + f16 write-back.
#pragma unroll
    for (int nb = 0; nb < 8; ++nb)
#pragma unroll
        for (int r = 0; r < 8; ++r) {
            float x = acc[nb][r];
            x = x > 0.0f ? x : 0.0f;
            Hw[(hi * 8 + r) * 128 + nb * 16 + row] = (_Float16)x;
        }
}

__global__ __launch_bounds__(TPB) void mlp_edge_score_kernel(
    const float* __restrict__ feat,
    const int*   __restrict__ src_idx,
    const int*   __restrict__ dst_idx,
    const float* __restrict__ W1,
    const float* __restrict__ W2,
    const float* __restrict__ W3,
    float* __restrict__ out,
    int nEdges) {
    __shared__ _Float16 Wt[128 * 128];        // 32 KB: transposed f16 weights [n][k]
    __shared__ _Float16 H[WAVES][16 * 128];   // 32 KB: per-wave 16-edge activation tile

    const int tid  = threadIdx.x;
    const int lane = tid & 31;
    const int wave = tid >> 5;
    const int row  = lane & 15;   // A-row / D-column index
    const int hi   = lane >> 4;   // half-wave selector for K striping

    _Float16* Hw   = &H[wave][0];
    const int ebase = blockIdx.x * (WAVES * 16) + wave * 16;

    // ---- phase 0: W1 -> LDS (f16, transposed) ----
    for (int i = tid; i < 128 * 128; i += TPB) {
        int k = i >> 7, n = i & 127;
        Wt[n * 128 + k] = (_Float16)W1[i];
    }
    __syncthreads();

    // ---- stage 0: gather + elementwise product -> H (f16) ----
    // Each lane covers 4 contiguous dims; feat rows are 512B -> b128 loads.
#pragma unroll 4
    for (int e = 0; e < 16; ++e) {
        int edge = ebase + e;
        int eg = edge < nEdges ? edge : nEdges - 1;
        const float4 fs = *(const float4*)(feat + (size_t)src_idx[eg] * 128 + lane * 4);
        const float4 fd = *(const float4*)(feat + (size_t)dst_idx[eg] * 128 + lane * 4);
        v4h p;
        p.x = (_Float16)(fs.x * fd.x);
        p.y = (_Float16)(fs.y * fd.y);
        p.z = (_Float16)(fs.z * fd.z);
        p.w = (_Float16)(fs.w * fd.w);
        *(v4h*)(Hw + e * 128 + lane * 4) = p;   // 8B aligned -> ds_store_b64
    }

    // ---- layer 1 ----
    layer128(Hw, Wt, row, hi);
    __syncthreads();

    // ---- phase 1: W2 -> LDS ----
    for (int i = tid; i < 128 * 128; i += TPB) {
        int k = i >> 7, n = i & 127;
        Wt[n * 128 + k] = (_Float16)W2[i];
    }
    __syncthreads();

    // ---- layer 2 ----
    layer128(Hw, Wt, row, hi);
    __syncthreads();

    // ---- phase 2: W3 (128x2) -> LDS as 16x128 transposed, zero padded ----
    for (int i = tid; i < 16 * 128; i += TPB) {
        int n = i >> 7, k = i & 127;
        Wt[i] = (n < 2) ? (_Float16)W3[k * 2 + n] : (_Float16)0.0f;
    }
    __syncthreads();

    // ---- layer 3: single 16-col N-tile, no relu, guarded f32 store ----
    {
        AF a[4];
        load_afrags(a, Hw + row * 128, hi);
        v8f acc = {};
        const _Float16* wrow = Wt + row * 128;
#pragma unroll
        for (int kb = 0; kb < 4; ++kb) {
            v16h b = load_bfrag(wrow, kb, hi);
            acc = __builtin_amdgcn_wmma_f32_16x16x32_f16(
                false, a[kb].v, false, b, (short)0, acc, false, false);
        }
        if (row < 2) {
#pragma unroll
            for (int r = 0; r < 8; ++r) {
                int edge = ebase + hi * 8 + r;
                if (edge < nEdges) out[(size_t)edge * 2 + row] = acc[r];
            }
        }
    }
}

extern "C" void kernel_launch(void* const* d_in, const int* in_sizes, int n_in,
                              void* d_out, int out_size, void* d_ws, size_t ws_size,
                              hipStream_t stream) {
    const float* feat = (const float*)d_in[0];
    const int*   src  = (const int*)d_in[1];
    const int*   dst  = (const int*)d_in[2];
    const float* W1   = (const float*)d_in[3];
    const float* W2   = (const float*)d_in[4];
    const float* W3   = (const float*)d_in[5];
    float* out = (float*)d_out;

    const int E = in_sizes[1];                // number of edges
    const int edgesPerBlock = WAVES * 16;     // 128
    const int blocks = (E + edgesPerBlock - 1) / edgesPerBlock;

    mlp_edge_score_kernel<<<blocks, TPB, 0, stream>>>(feat, src, dst, W1, W2, W3,
                                                      out, E);
}
